// NaiveFourierKANLayer_23733989278419
// MI455X (gfx1250) — compile-verified, matrix operand order checked by reading, not run on hardware
//
#include <hip/hip_runtime.h>

typedef __attribute__((ext_vector_type(16))) _Float16 v16h;
typedef __attribute__((ext_vector_type(8)))  _Float16 v8h;
typedef __attribute__((ext_vector_type(8)))  float    v8f;
typedef __attribute__((ext_vector_type(4)))  float    v4f;

#define B_SZ 8192
#define I_SZ 512
#define O_SZ 512
#define G_SZ 16
#define KTOT (I_SZ * G_SZ)        // per-trig-part K = 8192

#define BM 256
#define BN 128
#define KT 32
#define NTHREADS 512
#define NSTEPS (2 * (KTOT / KT))  // 512 total K-steps (cos part then sin part)

#define INV_2PI 0.15915494309189535f

__global__ __launch_bounds__(NTHREADS)
void fourier_kan_wmma_kernel(const float* __restrict__ x,
                             const float* __restrict__ coeffs,
                             const float* __restrict__ bias,
                             float* __restrict__ out)
{
    // Double-buffered feature tile [row][kk] and weight tile [n][kk], f16,
    // K-major per row so WMMA operand gathers are contiguous 16B LDS loads.
    __shared__ __align__(16) _Float16 Ft[2][BM * KT];   // 32 KB
    __shared__ __align__(16) _Float16 Wt[2][BN * KT];   // 16 KB

    const int tid  = threadIdx.x;
    const int lane = tid & 31;
    const int wave = tid >> 5;
    const int wm   = wave >> 1;            // 0..7 -> 32-row stripe
    const int wn   = wave & 1;             // 0..1 -> 64-col stripe
    const int row0 = blockIdx.x * BM;
    const int o0   = blockIdx.y * BN;

    // staging roles
    const int fr  = tid & (BM - 1);        // feature row 0..255
    const int fi  = tid >> 8;              // which of the 2 'i' columns
    const int wnn = tid >> 2;              // weight n 0..127
    const int wcc = (tid & 3) * 8;         // weight kk chunk 0,8,16,24

    v8f acc[8] = {};                       // [r*4 + c], r in 0..1, c in 0..3

    // per-lane K half-offset for the 16-bit WMMA A/B operand layout:
    // lanes 0-15: K = {0..7, 16..23}; lanes 16-31: K = {8..15, 24..31}
    const int hA      = (lane < 16) ? 0 : 8;
    const int aRowOff = (wm * 32 + (lane & 15)) * KT;    // A tile r adds r*16*KT
    const int bRowOff = (wn * 64 + (lane & 15)) * KT;    // B tile c adds c*16*KT

    // ---- issue global loads for step s ----
    auto issue_loads = [&](int s, float& xv, v4f& w0, v4f& w1) {
        const int ks = s & 255;
        const int p  = s >> 8;
        xv = x[(size_t)(row0 + fr) * I_SZ + ks * 2 + fi];
        const size_t base =
            (size_t)(p * O_SZ + o0 + wnn) * KTOT + (size_t)ks * KT + wcc;
        w0 = *(const v4f*)(coeffs + base);       // global_load_b128
        w1 = *(const v4f*)(coeffs + base + 4);   // global_load_b128
    };

    // ---- convert + store step s into LDS buffer `buf` ----
    auto stage = [&](int s, int buf, float xv, v4f w0, v4f w1) {
        const int   p  = s >> 8;                 // uniform scalar branch
        const float xr = xv * INV_2PI;           // v_sin/cos take revolutions
        v8h f0, f1;
        if (p == 0) {
            #pragma unroll
            for (int g = 0; g < 8; ++g)
                f0[g] = (_Float16)__builtin_amdgcn_cosf(xr * (float)(g + 1));
            #pragma unroll
            for (int g = 0; g < 8; ++g)
                f1[g] = (_Float16)__builtin_amdgcn_cosf(xr * (float)(g + 9));
        } else {
            #pragma unroll
            for (int g = 0; g < 8; ++g)
                f0[g] = (_Float16)__builtin_amdgcn_sinf(xr * (float)(g + 1));
            #pragma unroll
            for (int g = 0; g < 8; ++g)
                f1[g] = (_Float16)__builtin_amdgcn_sinf(xr * (float)(g + 9));
        }
        *(v8h*)&Ft[buf][fr * KT + fi * 16 + 0] = f0;   // ds_store_b128
        *(v8h*)&Ft[buf][fr * KT + fi * 16 + 8] = f1;   // ds_store_b128
        v8h wh;
        #pragma unroll
        for (int j = 0; j < 4; ++j) {
            wh[j]     = (_Float16)w0[j];
            wh[j + 4] = (_Float16)w1[j];
        }
        *(v8h*)&Wt[buf][wnn * KT + wcc] = wh;          // ds_store_b128
    };

    auto load16 = [&](const _Float16* rowp) -> v16h {
        v8h lo = *(const v8h*)(rowp + hA);
        v8h hi = *(const v8h*)(rowp + hA + 16);
        v16h r;
        #pragma unroll
        for (int j = 0; j < 8; ++j) { r[j] = lo[j]; r[j + 8] = hi[j]; }
        return r;
    };

    // prologue: stage step 0 into buffer 0; pre-issue loads for step 1
    float xv_p; v4f w0_p, w1_p;            // raw operands for step s+1
    {
        float xv; v4f w0, w1;
        issue_loads(0, xv, w0, w1);
        stage(0, 0, xv, w0, w1);
        issue_loads(1, xv_p, w0_p, w1_p);
    }
    __syncthreads();

    for (int s = 0; s < NSTEPS; ++s) {
        const int cur = s & 1;
        const int nxt = cur ^ 1;
        const int sn  = (s < NSTEPS - 1) ? s + 1 : s;      // step to stage
        const int sn2 = (s < NSTEPS - 2) ? s + 2 : NSTEPS - 1;  // step to load

        // 1) issue step s+2's global loads (a full iteration of latency cover)
        float xv_n; v4f w0_n, w1_n;
        issue_loads(sn2, xv_n, w0_n, w1_n);

        // 2) gather ALL operands (12 ds_load_b128), then 8 pipelined WMMAs
        v16h A0 = load16(&Ft[cur][aRowOff]);
        v16h A1 = load16(&Ft[cur][aRowOff + 16 * KT]);
        v16h Bt[4];
        #pragma unroll
        for (int c = 0; c < 4; ++c)
            Bt[c] = load16(&Wt[cur][bRowOff + c * 16 * KT]);

        #pragma unroll
        for (int c = 0; c < 4; ++c)
            acc[c] = __builtin_amdgcn_wmma_f32_16x16x32_f16(
                false, A0, false, Bt[c], (short)0, acc[c], false, false);
        #pragma unroll
        for (int c = 0; c < 4; ++c)
            acc[4 + c] = __builtin_amdgcn_wmma_f32_16x16x32_f16(
                false, A1, false, Bt[c], (short)0, acc[4 + c], false, false);

        // 3) stage step s+1 from registers loaded LAST iteration (no stall)
        stage(sn, nxt, xv_p, w0_p, w1_p);

        // Pin the shape: 3 VMEM reads -> 12 DS reads -> 8 WMMAs; everything
        // else (trig VALU, cvt, ds_stores) fills around the matrix pipe.
        __builtin_amdgcn_sched_group_barrier(0x020, 3,  0);  // VMEM read
        __builtin_amdgcn_sched_group_barrier(0x100, 12, 0);  // DS read
        __builtin_amdgcn_sched_group_barrier(0x008, 8,  0);  // MFMA/WMMA

        // 4) single barrier per K-step
        __syncthreads();

        // 5) rotate the raw-operand pipeline (copies vanish in the x2 unroll)
        xv_p = xv_n; w0_p = w0_n; w1_p = w1_n;
    }

    // ---- epilogue: add bias, store fp32 (C/D layout: VGPR j -> row j / j+8) ----
    #pragma unroll
    for (int r = 0; r < 2; ++r) {
        #pragma unroll
        for (int c = 0; c < 4; ++c) {
            const int   col  = o0 + wn * 64 + c * 16 + (lane & 15);
            const float bcol = bias[col];
            #pragma unroll
            for (int j = 0; j < 8; ++j) {
                const int row =
                    row0 + wm * 32 + r * 16 + ((lane < 16) ? j : j + 8);
                out[(size_t)row * O_SZ + col] = acc[r * 4 + c][j] + bcol;
            }
        }
    }
}

extern "C" void kernel_launch(void* const* d_in, const int* in_sizes, int n_in,
                              void* d_out, int out_size, void* d_ws, size_t ws_size,
                              hipStream_t stream) {
    const float* x      = (const float*)d_in[0];
    const float* coeffs = (const float*)d_in[1];
    const float* bias   = (const float*)d_in[2];
    float*       out    = (float*)d_out;

    dim3 grid(B_SZ / BM, O_SZ / BN);   // 32 x 4 = 128 blocks, 16 waves each
    fourier_kan_wmma_kernel<<<grid, NTHREADS, 0, stream>>>(x, coeffs, bias, out);
}
